// MHA_74594991997624
// MI455X (gfx1250) — matrix operand order repaired and checked by reading, not compiled, not run on hardware
//
#include <hip/hip_runtime.h>

typedef __bf16 bf16_t;
typedef __attribute__((ext_vector_type(8)))  __bf16 v8bf;
typedef __attribute__((ext_vector_type(16))) __bf16 v16bf;
typedef __attribute__((ext_vector_type(8)))  float  v8f;

union AFrag { v16bf f; v8bf h[2]; };

#if defined(__HIP_DEVICE_COMPILE__) && \
    __has_builtin(__builtin_amdgcn_global_load_async_to_lds_b128)
#define HAVE_ASYNC_LDS 1
typedef int v4i_gcc __attribute__((vector_size(16)));
typedef __attribute__((address_space(1))) v4i_gcc* async_gptr;
typedef __attribute__((address_space(3))) v4i_gcc* async_lptr;
#else
#define HAVE_ASYNC_LDS 0
#endif

static __device__ __forceinline__ bf16_t tobf(float x) {
  union { float f; unsigned u; } c; c.f = x;
  unsigned r = c.u + 0x7FFFu + ((c.u >> 16) & 1u);   // round-to-nearest-even
  unsigned short hs = (unsigned short)(r >> 16);
  bf16_t b;
  __builtin_memcpy(&b, &hs, 2);
  return b;
}

static __device__ __forceinline__ v8f vzero8() {
  v8f z = {0.f, 0.f, 0.f, 0.f, 0.f, 0.f, 0.f, 0.f};
  return z;
}

#define WMMA_BF16(A, Bm, C) \
  __builtin_amdgcn_wmma_f32_16x16x32_bf16(false, (A), false, (Bm), (short)0, (C), false, false)

// ---------------------------------------------------------------------------
// Kernel 1: fused Q/K/V projection.  X(8192x1024) x {Wq,Wk,Wv}(1024x64) per
// head, + bias, output bf16 into workspace as [h][s*B+b][64].
// Block: 256 threads (8 waves), covers 64 rows x 1 head (192 output cols).
// ---------------------------------------------------------------------------
__global__ __launch_bounds__(256) void qkv_proj_kernel(
    const float* __restrict__ X,
    const float* __restrict__ Wq, const float* __restrict__ bq,
    const float* __restrict__ Wk, const float* __restrict__ bk,
    const float* __restrict__ Wv, const float* __restrict__ bv,
    bf16_t* __restrict__ qws, bf16_t* __restrict__ kws, bf16_t* __restrict__ vws)
{
  __shared__ __attribute__((aligned(32))) bf16_t Xs[64][32];     // rows x K-chunk
  __shared__ __attribute__((aligned(32))) bf16_t Wts[192][32];   // [col][k] (transposed)

  const int tid  = threadIdx.x;
  const int lane = tid & 31;
  const int wid  = tid >> 5;           // wave 0..7
  const int half = (lane >> 4) & 1;
  const int lm   = lane & 15;
  const int rowBase = blockIdx.x * 64;
  const int h       = blockIdx.y;

  const int rt  = wid & 3;             // row tile 0..3
  const int ct0 = (wid >> 2) * 6;      // 6 col tiles per wave (of 12)

  v8f acc[6];
#pragma unroll
  for (int t = 0; t < 6; ++t) acc[t] = vzero8();

  const int smat = tid >> 6;           // 0..3 (3 = idle during W staging)
  const int sn   = tid & 63;
  const float* wsrc = (smat == 0) ? (Wq + (size_t)h * 1024 * 64)
                    : (smat == 1) ? (Wk + (size_t)h * 1024 * 64)
                                  : (Wv + (size_t)h * 1024 * 64);
  const int srow = tid >> 2, sseg = tid & 3;
  const float* gpX = X + (size_t)(rowBase + srow) * 1024 + sseg * 8;

  for (int k0 = 0; k0 < 1024; k0 += 32) {
    __syncthreads();
    // --- stage X strip (f32 -> bf16), 8 elems/thread
    {
      const float* gp = gpX + k0;
      float4 a = ((const float4*)gp)[0];
      float4 b = ((const float4*)gp)[1];
      bf16_t* dp = &Xs[srow][sseg * 8];
      dp[0] = tobf(a.x); dp[1] = tobf(a.y); dp[2] = tobf(a.z); dp[3] = tobf(a.w);
      dp[4] = tobf(b.x); dp[5] = tobf(b.y); dp[6] = tobf(b.z); dp[7] = tobf(b.w);
      if (k0 + 32 < 1024) __builtin_prefetch(gp + 32, 0, 1);   // global_prefetch_b8
    }
    // --- stage W transposed: Wts[mat*64+n][kk] = W[h][k0+kk][n]
    if (smat < 3) {
      const float* wp = wsrc + (size_t)k0 * 64 + sn;
#pragma unroll 4
      for (int kk = 0; kk < 32; ++kk)
        Wts[smat * 64 + sn][kk] = tobf(wp[(size_t)kk * 64]);
      if (k0 + 32 < 1024) __builtin_prefetch(wp + 32 * 64, 0, 1);
    }
    __syncthreads();

    // hoist all fragments, then issue WMMAs back-to-back
    AFrag a;
    a.h[0] = *(const v8bf*)&Xs[rt * 16 + lm][half * 16];
    a.h[1] = *(const v8bf*)&Xs[rt * 16 + lm][half * 16 + 8];
    v16bf bfr[6];
#pragma unroll
    for (int t = 0; t < 6; ++t)
      bfr[t] = *(const v16bf*)&Wts[(ct0 + t) * 16 + lm][half * 16];
#pragma unroll
    for (int t = 0; t < 6; ++t)
      acc[t] = WMMA_BF16(a.f, bfr[t], acc[t]);
  }

  // --- epilogue: bias + bf16 store into per-head q/k/v buffers
#pragma unroll
  for (int t = 0; t < 6; ++t) {
    int tile = ct0 + t;
    int mat  = tile >> 2;                       // 0=q,1=k,2=v (uniform per tile)
    int coln = ((tile & 3) << 4) + lm;          // 0..63 within head
    const float* bptr = (mat == 0) ? bq : (mat == 1) ? bk : bv;
    bf16_t* op = ((mat == 0) ? qws : (mat == 1) ? kws : vws)
                 + (size_t)h * 8192 * 64 + coln;
    float bb = bptr[h * 64 + coln];
#pragma unroll
    for (int j = 0; j < 8; ++j) {
      int row = rowBase + rt * 16 + j + half * 8;
      op[(size_t)row * 64] = tobf(acc[t][j] + bb);
    }
  }
}

// ---------------------------------------------------------------------------
// Kernel 2: causal flash attention per (h, b).  Block = 128 threads (4 waves),
// 64 query rows; kv tiles of 32 keys.  Scores & P.V via bf16 WMMA, online
// softmax with __shfl_xor reductions inside 16-lane halves (wave32).
// ---------------------------------------------------------------------------
__global__ __launch_bounds__(128) void attn_kernel(
    const bf16_t* __restrict__ qws, const bf16_t* __restrict__ kws,
    const bf16_t* __restrict__ vws, bf16_t* __restrict__ yws)
{
  __shared__ __attribute__((aligned(32))) bf16_t Vt[64][32];      // [dv][key]
  __shared__ __attribute__((aligned(32))) bf16_t Ps[4][16][32];   // per-wave P

  const int tid  = threadIdx.x;
  const int lane = tid & 31;
  const int w    = tid >> 5;           // 0..3
  const int half = (lane >> 4) & 1;
  const int lm   = lane & 15;

  const int qtile = blockIdx.x;        // 0..15
  const int b     = blockIdx.y;        // 0..7
  const int h     = blockIdx.z;        // 0..15
  const int qbase = qtile * 64;
  const int swave = qbase + w * 16;

  const size_t headoff = (size_t)h * 8192 * 64;

  // Q fragments (dk=64 -> two K=32 chunks), rows = swave + lm
  const bf16_t* qrow = qws + headoff + (size_t)((swave + lm) * 8 + b) * 64;
  AFrag qf[2];
#pragma unroll
  for (int c = 0; c < 2; ++c) {
    qf[c].h[0] = *(const v8bf*)&qrow[c * 32 + half * 16];
    qf[c].h[1] = *(const v8bf*)&qrow[c * 32 + half * 16 + 8];
  }

  v8f O[4];
#pragma unroll
  for (int t = 0; t < 4; ++t) O[t] = vzero8();
  float m[8], l[8];
#pragma unroll
  for (int j = 0; j < 8; ++j) { m[j] = -1e30f; l[j] = 0.f; }

  const int nkv = qtile * 2 + 2;       // causal: keys < qbase+64
  for (int kt = 0; kt < nkv; ++kt) {
    const int kcol0 = kt * 32;

    __syncthreads();
    // stage V^T tile (32 keys x 64 dv) -> Vt[dv][key]
#pragma unroll
    for (int i = 0; i < 16; ++i) {
      int idx = i * 128 + tid;                 // 0..2047
      int key = idx >> 6;
      int dv  = idx & 63;
      Vt[dv][key] = vws[headoff + (size_t)((kcol0 + key) * 8 + b) * 64 + dv];
    }
    __syncthreads();

    // scores: Q(16x64) x K^T -> two 16x16 tiles (fragments hoisted)
    const bf16_t* krow0 =
        kws + headoff + (size_t)((kcol0 + lm) * 8 + b) * 64;
    const bf16_t* krow1 =
        kws + headoff + (size_t)((kcol0 + 16 + lm) * 8 + b) * 64;
    v16bf k00 = *(const v16bf*)&krow0[half * 16];
    v16bf k01 = *(const v16bf*)&krow0[32 + half * 16];
    v16bf k10 = *(const v16bf*)&krow1[half * 16];
    v16bf k11 = *(const v16bf*)&krow1[32 + half * 16];
    v8f st[2];
    st[0] = vzero8();
    st[0] = WMMA_BF16(qf[0].f, k00, st[0]);
    st[0] = WMMA_BF16(qf[1].f, k01, st[0]);
    st[1] = vzero8();
    st[1] = WMMA_BF16(qf[0].f, k10, st[1]);
    st[1] = WMMA_BF16(qf[1].f, k11, st[1]);

    // scale + causal mask + online softmax (row = VGPR j, lane half picks j/j+8)
#pragma unroll
    for (int j = 0; j < 8; ++j) {
      const int sq = swave + j + half * 8;
      float s0 = st[0][j] * 0.125f;            // 1/sqrt(64)
      float s1 = st[1][j] * 0.125f;
      if (kcol0 + lm > sq)      s0 = -1e30f;
      if (kcol0 + 16 + lm > sq) s1 = -1e30f;
      float vmax = fmaxf(s0, s1);
#pragma unroll
      for (int msk = 1; msk < 16; msk <<= 1)
        vmax = fmaxf(vmax, __shfl_xor(vmax, msk, 32));
      float mnew = fmaxf(m[j], vmax);
      float corr = __expf(m[j] - mnew);
      float p0 = __expf(s0 - mnew);
      float p1 = __expf(s1 - mnew);
      float ps = p0 + p1;
#pragma unroll
      for (int msk = 1; msk < 16; msk <<= 1)
        ps += __shfl_xor(ps, msk, 32);
      l[j] = l[j] * corr + ps;
      m[j] = mnew;
#pragma unroll
      for (int t = 0; t < 4; ++t) O[t][j] *= corr;
      int r = j + half * 8;
      Ps[w][r][lm]      = tobf(p0);            // D-layout -> LDS
      Ps[w][r][16 + lm] = tobf(p1);
    }

    // P(16x32) x V(32x64): reload P in A-layout (same-wave LDS is in-order)
    AFrag pf;
    pf.h[0] = *(const v8bf*)&Ps[w][lm][half * 16];
    pf.h[1] = *(const v8bf*)&Ps[w][lm][half * 16 + 8];
    v16bf vf[4];
#pragma unroll
    for (int t = 0; t < 4; ++t)
      vf[t] = *(const v16bf*)&Vt[t * 16 + lm][half * 16];
#pragma unroll
    for (int t = 0; t < 4; ++t)
      O[t] = WMMA_BF16(pf.f, vf[t], O[t]);
  }

  // normalize + store concat(head) row into yws (bf16)
#pragma unroll
  for (int j = 0; j < 8; ++j) {
    float inv = 1.0f / l[j];
    int sq = swave + j + half * 8;
    bf16_t* yp = yws + (size_t)(sq * 8 + b) * 1024 + h * 64;
#pragma unroll
    for (int t = 0; t < 4; ++t)
      yp[t * 16 + lm] = tobf(O[t][j] * inv);
  }
}

// ---------------------------------------------------------------------------
// Kernel 3: output projection  Y(8192x1024) x Wo(1024x1024) + bo -> f32 out.
// Block: 256 threads (8 waves), 64 rows x 128 cols.  X tile staged with the
// gfx1250 async global->LDS path when available (ASYNCcnt), else plain copy.
// ---------------------------------------------------------------------------
__global__ __launch_bounds__(256) void out_proj_kernel(
    const bf16_t* __restrict__ yws, const float* __restrict__ Wo,
    const float* __restrict__ bo, float* __restrict__ out)
{
  __shared__ __attribute__((aligned(32))) bf16_t Xs[64][32];
  __shared__ __attribute__((aligned(32))) bf16_t Wts[128][32];   // [col][k]

  const int tid  = threadIdx.x;
  const int lane = tid & 31;
  const int wid  = tid >> 5;
  const int half = (lane >> 4) & 1;
  const int lm   = lane & 15;
  const int rowBase = blockIdx.x * 64;
  const int colBase = blockIdx.y * 128;

  const int rt  = wid & 3;
  const int ct0 = (wid >> 2) * 4;

  v8f acc[4];
#pragma unroll
  for (int t = 0; t < 4; ++t) acc[t] = vzero8();

  const int srow = tid >> 2, sseg = tid & 3;

  for (int k0 = 0; k0 < 1024; k0 += 32) {
    __syncthreads();
    // --- stage X strip (already bf16): async DMA to LDS when available
    {
      const bf16_t* gsrc =
          yws + (size_t)(rowBase + srow) * 1024 + k0 + sseg * 8;
#if HAVE_ASYNC_LDS
      __builtin_amdgcn_global_load_async_to_lds_b128(
          (async_gptr)gsrc, (async_lptr)(&Xs[srow][sseg * 8]), 0, 0);
#else
      *(v8bf*)&Xs[srow][sseg * 8] = *(const v8bf*)gsrc;
#endif
      if (k0 + 32 < 1024) __builtin_prefetch(gsrc + 32, 0, 1);
    }
    // --- stage W transposed (f32 -> bf16)
#pragma unroll
    for (int i = 0; i < 16; ++i) {
      int idx = i * 256 + tid;                 // 0..4095
      int col = idx & 127;
      int kk  = idx >> 7;
      Wts[col][kk] = tobf(Wo[(size_t)(k0 + kk) * 1024 + colBase + col]);
    }
    if (k0 + 32 < 1024)
      __builtin_prefetch(Wo + (size_t)(k0 + 32) * 1024 + colBase + (tid & 127), 0, 1);
#if HAVE_ASYNC_LDS
    asm volatile("s_wait_asynccnt 0x0" ::: "memory");
#endif
    __syncthreads();

    AFrag a;
    a.h[0] = *(const v8bf*)&Xs[rt * 16 + lm][half * 16];
    a.h[1] = *(const v8bf*)&Xs[rt * 16 + lm][half * 16 + 8];
    v16bf bfr[4];
#pragma unroll
    for (int t = 0; t < 4; ++t)
      bfr[t] = *(const v16bf*)&Wts[(ct0 + t) * 16 + lm][half * 16];
#pragma unroll
    for (int t = 0; t < 4; ++t)
      acc[t] = WMMA_BF16(a.f, bfr[t], acc[t]);
  }

#pragma unroll
  for (int t = 0; t < 4; ++t) {
    int col = colBase + (ct0 + t) * 16 + lm;
    float bb = bo[col];
#pragma unroll
    for (int j = 0; j < 8; ++j) {
      int row = rowBase + rt * 16 + j + half * 8;
      out[(size_t)row * 1024 + col] = acc[t][j] + bb;
    }
  }
}

// ---------------------------------------------------------------------------
extern "C" void kernel_launch(void* const* d_in, const int* in_sizes, int n_in,
                              void* d_out, int out_size, void* d_ws, size_t ws_size,
                              hipStream_t stream) {
  const float* Q  = (const float*)d_in[0];
  const float* Wq = (const float*)d_in[1];
  const float* bq = (const float*)d_in[2];
  const float* Wk = (const float*)d_in[3];
  const float* bk = (const float*)d_in[4];
  const float* Wv = (const float*)d_in[5];
  const float* bv = (const float*)d_in[6];
  const float* Wo = (const float*)d_in[7];
  const float* bo = (const float*)d_in[8];
  float* out = (float*)d_out;

  const size_t HNE = (size_t)16 * 8192 * 64;   // elems per q/k/v buffer (bf16)
  bf16_t* qws = (bf16_t*)d_ws;
  bf16_t* kws = qws + HNE;
  bf16_t* vws = kws + HNE;
  bf16_t* yws = vws + HNE;                     // 8192 x 1024 bf16

  qkv_proj_kernel<<<dim3(128, 16), 256, 0, stream>>>(Q, Wq, bq, Wk, bk, Wv, bv,
                                                     qws, kws, vws);
  attn_kernel<<<dim3(16, 8, 16), 128, 0, stream>>>(qws, kws, vws, yws);
  out_proj_kernel<<<dim3(128, 8), 256, 0, stream>>>(yws, Wo, bo, out);
}
